// RWKV_Tmix_x070_50663434224468
// MI455X (gfx1250) — matrix operand order repaired and checked by reading, not compiled, not run on hardware
//
#include <hip/hip_runtime.h>
#include <hip/hip_bf16.h>
#include <cstdint>

// ---------------- constants (reference shapes) ----------------
constexpr int Bsz = 4, T = 2048, C = 1024, H = 16, NH = 64;
constexpr int M   = Bsz * T;                 // 8192 rows
constexpr int Dw = 64, Da = 64, Dv = 32, Dg = 128;
constexpr float EPS_GN = 1e-5f * 64.0f;      // eps * head_size_divisor^2 = 6.4e-4

// ---------------- CDNA5 WMMA types ----------------
typedef __attribute__((ext_vector_type(16))) __bf16 v16bf;
typedef __attribute__((ext_vector_type(8)))  float  v8f;

union Frag16 { v16bf v; unsigned short u[16]; uint4 q[2]; };

#define AS1 __attribute__((address_space(1)))
#define AS3 __attribute__((address_space(3)))

// exact pointee types expected by the async builtins (per hipcc diagnostics)
typedef int vsi4 __attribute__((vector_size(16)));

#if __has_builtin(__builtin_amdgcn_global_load_async_to_lds_b128) && \
    __has_builtin(__builtin_amdgcn_global_load_async_to_lds_b32)
#define HAVE_ASYNC_LDS 1
#else
#define HAVE_ASYNC_LDS 0
#endif

// async global->LDS (ASYNCcnt-tracked) with safe fallback
static __device__ __forceinline__ void async_b128(const void* g, void* l) {
#if HAVE_ASYNC_LDS
    __builtin_amdgcn_global_load_async_to_lds_b128((AS1 vsi4*)g, (AS3 vsi4*)l, 0, 0);
#else
    *reinterpret_cast<uint4*>(l) = *reinterpret_cast<const uint4*>(g);
#endif
}
static __device__ __forceinline__ void async_b32(const void* g, void* l) {
#if HAVE_ASYNC_LDS
    __builtin_amdgcn_global_load_async_to_lds_b32((AS1 int*)g, (AS3 int*)l, 0, 0);
#else
    *reinterpret_cast<unsigned*>(l) = *reinterpret_cast<const unsigned*>(g);
#endif
}
static __device__ __forceinline__ void wait_async0() {
#if HAVE_ASYNC_LDS
#if __has_builtin(__builtin_amdgcn_s_wait_asynccnt)
    __builtin_amdgcn_s_wait_asynccnt(0);
#else
    asm volatile("s_wait_asynccnt 0x0" ::: "memory");
#endif
#endif
}

static __device__ __forceinline__ unsigned short f2bf(float f) {
    unsigned u = __builtin_bit_cast(unsigned, f);
    unsigned r = u + 0x7FFFu + ((u >> 16) & 1u);   // round-to-nearest-even
    return (unsigned short)(r >> 16);
}
static __device__ __forceinline__ float sigmoidf_(float x) {
    return 1.0f / (1.0f + expf(-x));
}

// ---------------- GEMM: C[M,N] = A[M,K](bf16) * Bt[N,K](bf16)^T, f32 accum ----------------
// A 16-row tile is shared by all WAVES waves of the block: staged in LDS via
// double-buffered async global->LDS loads; B fragments stream from global/L2.
template<int WAVES>
__global__ void __launch_bounds__(32 * WAVES)
gemm_bf16_nt(const unsigned short* __restrict__ A,
             const unsigned short* __restrict__ Bt,
             float* __restrict__ Cmat,
             int Mm, int Nn, int Kk)
{
    __shared__ __align__(16) unsigned short atile[2][16 * 32];   // 2 x 1KB
    const int tid  = threadIdx.x;
    const int lane = tid & 31;
    const int wave = tid >> 5;
    const int tm   = blockIdx.x;
    const int tn   = blockIdx.y * WAVES + wave;
    const int hi   = lane >> 4;
    const int l15  = lane & 15;

    const unsigned short* brow  = Bt + (size_t)(tn * 16 + l15) * Kk;
    const unsigned short* abase = A + (size_t)(tm * 16) * Kk;

    // async A-tile loader: 64 chunks of 16B; threads 0..63 (waves 0,1) issue
    auto loadA = [&](int k0, int bufIdx) {
        if (tid < 64) {
            const int row = tid >> 2, seg = tid & 3;
            async_b128(abase + (size_t)row * Kk + k0 + seg * 8,
                       &atile[bufIdx][row * 32 + seg * 8]);
        }
    };

    loadA(0, 0);
    wait_async0();
    __syncthreads();

    v8f acc = {};
    const int nIter = Kk >> 5;
    for (int it = 0; it < nIter; ++it) {
        const int cur = it & 1;
        if (it + 1 < nIter) loadA((it + 1) * 32, cur ^ 1);

        Frag16 a, b;
        const unsigned short* as = &atile[cur][l15 * 32];
        a.q[0] = *reinterpret_cast<const uint4*>(as + hi * 8);
        a.q[1] = *reinterpret_cast<const uint4*>(as + hi * 8 + 16);
        const int kb = it * 32 + hi * 8;
        b.q[0] = *reinterpret_cast<const uint4*>(brow + kb);
        b.q[1] = *reinterpret_cast<const uint4*>(brow + kb + 16);
        if (kb + 160 <= Kk)
            __builtin_prefetch(brow + kb + 128, 0, 1);   // global_prefetch_b8

        acc = __builtin_amdgcn_wmma_f32_16x16x32_bf16(
                  false, a.v, false, b.v, (short)0, acc, false, false);

        wait_async0();
        __syncthreads();
    }

    float* cbase = Cmat + (size_t)(tm * 16 + hi * 8) * Nn + tn * 16 + l15;
#pragma unroll
    for (int r = 0; r < 8; ++r)
        cbase[(size_t)r * Nn] = acc[r];
    (void)Mm;
}

// ---------------- weight prep ----------------
__global__ void convert_kernel(const float* __restrict__ in, unsigned short* __restrict__ out, int n) {
    int i = blockIdx.x * blockDim.x + threadIdx.x;
    if (i < n) out[i] = f2bf(in[i]);
}
// in[Kdim][Ndim] (f32) -> out[Ndim][Kdim] (bf16)
__global__ void transpose_conv_kernel(const float* __restrict__ in, unsigned short* __restrict__ out,
                                      int Kdim, int Ndim) {
    int i = blockIdx.x * blockDim.x + threadIdx.x;
    if (i >= Kdim * Ndim) return;
    int k = i / Ndim, n = i % Ndim;
    out[(size_t)n * Kdim + k] = f2bf(in[i]);
}
// mode: 0 = copy, 1 = tanh, 2 = sigmoid
__global__ void act_conv_kernel(const float* __restrict__ in, unsigned short* __restrict__ out,
                                int n, int mode) {
    int i = blockIdx.x * blockDim.x + threadIdx.x;
    if (i >= n) return;
    float v = in[i];
    if (mode == 1) v = tanhf(v);
    else if (mode == 2) v = sigmoidf_(v);
    out[i] = f2bf(v);
}

// ---------------- token mix: six bf16 activation matrices ----------------
__global__ void mix_kernel(const float* __restrict__ x,
                           const float* __restrict__ cr, const float* __restrict__ cw,
                           const float* __restrict__ ck, const float* __restrict__ cv,
                           const float* __restrict__ ca, const float* __restrict__ cg,
                           unsigned short* __restrict__ oxr, unsigned short* __restrict__ oxw,
                           unsigned short* __restrict__ oxk, unsigned short* __restrict__ oxv,
                           unsigned short* __restrict__ oxa, unsigned short* __restrict__ oxg)
{
    size_t idx = (size_t)blockIdx.x * blockDim.x + threadIdx.x;
    if (idx >= (size_t)M * C) return;
    int c = (int)(idx % C);
    int m = (int)(idx / C);
    int t = m % T;
    float xc = x[idx];
    float xx = (t > 0 ? x[idx - C] : 0.0f) - xc;
    oxr[idx] = f2bf(xc + xx * cr[c]);
    oxw[idx] = f2bf(xc + xx * cw[c]);
    oxk[idx] = f2bf(xc + xx * ck[c]);
    oxv[idx] = f2bf(xc + xx * cv[c]);
    oxa[idx] = f2bf(xc + xx * ca[c]);
    oxg[idx] = f2bf(xc + xx * cg[c]);
}

// ---------------- LDS 64-thread reduction ----------------
static __device__ __forceinline__ float blockReduce64(float v, float* red, int i) {
    red[i] = v; __syncthreads();
#pragma unroll
    for (int s = 32; s > 0; s >>= 1) { if (i < s) red[i] += red[i + s]; __syncthreads(); }
    float r = red[0]; __syncthreads();
    return r;
}

// ---------------- post1: decay, kk-norm, k update, v mix; one block per (m,h) ----------------
__global__ void __launch_bounds__(64)
post1_kernel(float* __restrict__ kbuf,  // in: k = xk@Wk.T, out: updated k
             float* __restrict__ vbuf,  // in: vraw = xv@Wv.T, out: mixed v
             float* __restrict__ w2o,   // in: tanh(xw@w1)@w2, out: decay
             float* __restrict__ a2o,   // in: (xa@a1)@a2,     out: aneg = -kk_norm
             float* __restrict__ v2o,   // in: (xv@v1)@v2,     out: bb   = kk_norm * a_sig
             const float* __restrict__ v_first,
             const float* __restrict__ k_k, const float* __restrict__ k_a,
             const float* __restrict__ a0,  const float* __restrict__ v0,
             const float* __restrict__ w0)
{
    __shared__ float red[64];
    int mh = blockIdx.x;
    int m = mh / H, h = mh % H, i = threadIdx.x;
    int c = h * NH + i;
    size_t idx = (size_t)m * C + c;

    float kv = kbuf[idx];
    float kk = kv * k_k[c];
    float ss = blockReduce64(kk * kk, red, i);
    float kkn = kk / fmaxf(sqrtf(ss), 1e-12f);

    float asig = sigmoidf_(a0[c] + a2o[idx]);
    float knew = kv * (1.0f + (asig - 1.0f) * k_a[c]);

    float wv = w0[c] + w2o[idx];
    float w  = -log1pf(expf(-wv)) - 0.5f;      // -softplus(-wv) - 0.5
    float dec = expf(-expf(w));

    float sv   = sigmoidf_(v0[c] + v2o[idx]);
    float vraw = vbuf[idx];
    float vnew = vraw + (v_first[idx] - vraw) * sv;

    kbuf[idx] = knew;
    vbuf[idx] = vnew;
    w2o[idx]  = dec;
    a2o[idx]  = -kkn;
    v2o[idx]  = kkn * asig;
}

// ---------------- WKV7 scan: one block (64 threads) per (b,h); S in VGPRs ----------------
// Step vectors double-buffered in LDS; t+1 fetched via async global->LDS loads
// overlapping the ~300 VALU ops of step t.
__global__ void __launch_bounds__(64)
wkv_kernel(const float* __restrict__ r, const float* __restrict__ dec,
           const float* __restrict__ k, const float* __restrict__ v,
           const float* __restrict__ an, const float* __restrict__ bn,
           float* __restrict__ y)
{
    __shared__ float buf[2][6][64];
    int bh = blockIdx.x;
    int b = bh / H, h = bh % H, i = threadIdx.x;
    float S[64];
#pragma unroll
    for (int j = 0; j < 64; ++j) S[j] = 0.0f;

    size_t base = (size_t)b * T * C + h * NH;

    // prologue: stage t = 0
    async_b32(r  + base + i, &buf[0][0][i]);
    async_b32(dec+ base + i, &buf[0][1][i]);
    async_b32(k  + base + i, &buf[0][2][i]);
    async_b32(v  + base + i, &buf[0][3][i]);
    async_b32(an + base + i, &buf[0][4][i]);
    async_b32(bn + base + i, &buf[0][5][i]);
    wait_async0();
    __syncthreads();

    for (int t = 0; t < T; ++t, base += C) {
        const int cur = t & 1, nxt = cur ^ 1;
        if (t + 1 < T) {
            size_t nb = base + C;
            async_b32(r  + nb + i, &buf[nxt][0][i]);
            async_b32(dec+ nb + i, &buf[nxt][1][i]);
            async_b32(k  + nb + i, &buf[nxt][2][i]);
            async_b32(v  + nb + i, &buf[nxt][3][i]);
            async_b32(an + nb + i, &buf[nxt][4][i]);
            async_b32(bn + nb + i, &buf[nxt][5][i]);
        }

        const float* lr = buf[cur][0];
        const float* ld = buf[cur][1];
        const float* lk = buf[cur][2];
        const float* lv = buf[cur][3];
        const float* la = buf[cur][4];
        const float* lb = buf[cur][5];

        float sa = 0.0f;
#pragma unroll
        for (int j = 0; j < 64; ++j) sa += S[j] * la[j];
        float vi = lv[i];
        float yv = 0.0f;
#pragma unroll
        for (int j = 0; j < 64; ++j) {
            float s = S[j] * ld[j] + sa * lb[j] + vi * lk[j];
            S[j] = s;
            yv += s * lr[j];
        }
        y[base + i] = yv;

        wait_async0();
        __syncthreads();
    }
}

// ---------------- post2: GroupNorm + rk residual + gate -> bf16 for final GEMM ----------------
__global__ void __launch_bounds__(64)
post2_kernel(const float* __restrict__ y, const float* __restrict__ r,
             const float* __restrict__ k, const float* __restrict__ v,
             const float* __restrict__ g, const float* __restrict__ r_k,
             const float* __restrict__ ln_g, const float* __restrict__ ln_b,
             unsigned short* __restrict__ yg)
{
    __shared__ float red[64];
    int mh = blockIdx.x;
    int m = mh / H, h = mh % H, i = threadIdx.x;
    int c = h * NH + i;
    size_t idx = (size_t)m * C + c;

    float yv = y[idx];
    float mu = blockReduce64(yv, red, i) * (1.0f / 64.0f);
    float d  = yv - mu;
    float var = blockReduce64(d * d, red, i) * (1.0f / 64.0f);
    float xn = d * rsqrtf(var + EPS_GN);
    float yln = xn * ln_g[c] + ln_b[c];

    float rk = blockReduce64(r[idx] * k[idx] * r_k[h * NH + i], red, i);
    float out = yln + rk * v[idx];
    yg[idx] = f2bf(out * g[idx]);
}

// ---------------- launch ----------------
extern "C" void kernel_launch(void* const* d_in, const int* in_sizes, int n_in,
                              void* d_out, int out_size, void* d_ws, size_t ws_size,
                              hipStream_t stream)
{
    const float* x       = (const float*)d_in[0];
    const float* v_first = (const float*)d_in[1];
    const float* x_r = (const float*)d_in[2];
    const float* x_w = (const float*)d_in[3];
    const float* x_k = (const float*)d_in[4];
    const float* x_v = (const float*)d_in[5];
    const float* x_a = (const float*)d_in[6];
    const float* x_g = (const float*)d_in[7];
    const float* w0  = (const float*)d_in[8];
    const float* w1  = (const float*)d_in[9];
    const float* w2  = (const float*)d_in[10];
    const float* a0  = (const float*)d_in[11];
    const float* a1  = (const float*)d_in[12];
    const float* a2  = (const float*)d_in[13];
    const float* v0  = (const float*)d_in[14];
    const float* v1  = (const float*)d_in[15];
    const float* v2  = (const float*)d_in[16];
    const float* g1  = (const float*)d_in[17];
    const float* g2  = (const float*)d_in[18];
    const float* k_k = (const float*)d_in[19];
    const float* k_a = (const float*)d_in[20];
    const float* r_k = (const float*)d_in[21];
    const float* Wr  = (const float*)d_in[22];
    const float* Wk  = (const float*)d_in[23];
    const float* Wv  = (const float*)d_in[24];
    const float* Wo  = (const float*)d_in[25];
    const float* ln_g = (const float*)d_in[26];
    const float* ln_b = (const float*)d_in[27];

    // bump allocator over d_ws
    char* ws = (char*)d_ws;
    size_t off = 0;
    auto alloc = [&](size_t bytes) -> void* {
        void* p = ws + off;
        off = (off + bytes + 255) & ~(size_t)255;
        return p;
    };
    const size_t MC = (size_t)M * C;

    unsigned short* xr_b = (unsigned short*)alloc(MC * 2);
    unsigned short* xw_b = (unsigned short*)alloc(MC * 2);
    unsigned short* xk_b = (unsigned short*)alloc(MC * 2);
    unsigned short* xv_b = (unsigned short*)alloc(MC * 2);
    unsigned short* xa_b = (unsigned short*)alloc(MC * 2);
    unsigned short* xg_b = (unsigned short*)alloc(MC * 2);
    unsigned short* yg_b = (unsigned short*)alloc(MC * 2);

    unsigned short* Wr_b = (unsigned short*)alloc((size_t)C * C * 2);
    unsigned short* Wk_b = (unsigned short*)alloc((size_t)C * C * 2);
    unsigned short* Wv_b = (unsigned short*)alloc((size_t)C * C * 2);
    unsigned short* Wo_b = (unsigned short*)alloc((size_t)C * C * 2);
    unsigned short* w1t = (unsigned short*)alloc((size_t)Dw * C * 2);
    unsigned short* w2t = (unsigned short*)alloc((size_t)C * Dw * 2);
    unsigned short* a1t = (unsigned short*)alloc((size_t)Da * C * 2);
    unsigned short* a2t = (unsigned short*)alloc((size_t)C * Da * 2);
    unsigned short* v1t = (unsigned short*)alloc((size_t)Dv * C * 2);
    unsigned short* v2t = (unsigned short*)alloc((size_t)C * Dv * 2);
    unsigned short* g1t = (unsigned short*)alloc((size_t)Dg * C * 2);
    unsigned short* g2t = (unsigned short*)alloc((size_t)C * Dg * 2);

    float* rbuf = (float*)alloc(MC * 4);
    float* kbuf = (float*)alloc(MC * 4);
    float* vbuf = (float*)alloc(MC * 4);
    float* w2o  = (float*)alloc(MC * 4);   // -> decay
    float* a2o  = (float*)alloc(MC * 4);   // -> aneg
    float* v2o  = (float*)alloc(MC * 4);   // -> bb
    float* ybuf = (float*)alloc(MC * 4);
    float* gbuf = (float*)alloc(MC * 4);

    float* hw = (float*)alloc((size_t)M * Dw * 4);
    float* ha = (float*)alloc((size_t)M * Da * 4);
    float* hv = (float*)alloc((size_t)M * Dv * 4);
    float* hg = (float*)alloc((size_t)M * Dg * 4);
    unsigned short* hw_b = (unsigned short*)alloc((size_t)M * Dw * 2);
    unsigned short* ha_b = (unsigned short*)alloc((size_t)M * Da * 2);
    unsigned short* hv_b = (unsigned short*)alloc((size_t)M * Dv * 2);
    unsigned short* hg_b = (unsigned short*)alloc((size_t)M * Dg * 2);
    (void)ws_size; (void)n_in; (void)in_sizes; (void)out_size;

    const int TB = 256;
    auto gridFor = [&](size_t n) { return (unsigned)((n + TB - 1) / TB); };

    // ---- weight prep ----
    convert_kernel<<<gridFor((size_t)C * C), TB, 0, stream>>>(Wr, Wr_b, C * C);
    convert_kernel<<<gridFor((size_t)C * C), TB, 0, stream>>>(Wk, Wk_b, C * C);
    convert_kernel<<<gridFor((size_t)C * C), TB, 0, stream>>>(Wv, Wv_b, C * C);
    convert_kernel<<<gridFor((size_t)C * C), TB, 0, stream>>>(Wo, Wo_b, C * C);
    transpose_conv_kernel<<<gridFor((size_t)C * Dw), TB, 0, stream>>>(w1, w1t, C, Dw);
    transpose_conv_kernel<<<gridFor((size_t)Dw * C), TB, 0, stream>>>(w2, w2t, Dw, C);
    transpose_conv_kernel<<<gridFor((size_t)C * Da), TB, 0, stream>>>(a1, a1t, C, Da);
    transpose_conv_kernel<<<gridFor((size_t)Da * C), TB, 0, stream>>>(a2, a2t, Da, C);
    transpose_conv_kernel<<<gridFor((size_t)C * Dv), TB, 0, stream>>>(v1, v1t, C, Dv);
    transpose_conv_kernel<<<gridFor((size_t)Dv * C), TB, 0, stream>>>(v2, v2t, Dv, C);
    transpose_conv_kernel<<<gridFor((size_t)C * Dg), TB, 0, stream>>>(g1, g1t, C, Dg);
    transpose_conv_kernel<<<gridFor((size_t)Dg * C), TB, 0, stream>>>(g2, g2t, Dg, C);

    // ---- token mix ----
    mix_kernel<<<gridFor(MC), TB, 0, stream>>>(x, x_r, x_w, x_k, x_v, x_a, x_g,
                                               xr_b, xw_b, xk_b, xv_b, xa_b, xg_b);

    // ---- stage-1 GEMMs (WMMA + async LDS staging) ----
    gemm_bf16_nt<8><<<dim3(M / 16, C / 128), 256, 0, stream>>>(xr_b, Wr_b, rbuf, M, C, C);
    gemm_bf16_nt<8><<<dim3(M / 16, C / 128), 256, 0, stream>>>(xk_b, Wk_b, kbuf, M, C, C);
    gemm_bf16_nt<8><<<dim3(M / 16, C / 128), 256, 0, stream>>>(xv_b, Wv_b, vbuf, M, C, C);
    gemm_bf16_nt<4><<<dim3(M / 16, 1), 128, 0, stream>>>(xw_b, w1t, hw, M, Dw, C);
    gemm_bf16_nt<4><<<dim3(M / 16, 1), 128, 0, stream>>>(xa_b, a1t, ha, M, Da, C);
    gemm_bf16_nt<2><<<dim3(M / 16, 1),  64, 0, stream>>>(xv_b, v1t, hv, M, Dv, C);
    gemm_bf16_nt<8><<<dim3(M / 16, 1), 256, 0, stream>>>(xg_b, g1t, hg, M, Dg, C);

    // ---- activations between low-rank stages ----
    act_conv_kernel<<<gridFor((size_t)M * Dw), TB, 0, stream>>>(hw, hw_b, M * Dw, 1); // tanh
    act_conv_kernel<<<gridFor((size_t)M * Da), TB, 0, stream>>>(ha, ha_b, M * Da, 0); // copy
    act_conv_kernel<<<gridFor((size_t)M * Dv), TB, 0, stream>>>(hv, hv_b, M * Dv, 0); // copy
    act_conv_kernel<<<gridFor((size_t)M * Dg), TB, 0, stream>>>(hg, hg_b, M * Dg, 2); // sigmoid

    // ---- stage-2 GEMMs (WMMA) ----
    gemm_bf16_nt<8><<<dim3(M / 16, C / 128), 256, 0, stream>>>(hw_b, w2t, w2o, M, C, Dw);
    gemm_bf16_nt<8><<<dim3(M / 16, C / 128), 256, 0, stream>>>(ha_b, a2t, a2o, M, C, Da);
    gemm_bf16_nt<8><<<dim3(M / 16, C / 128), 256, 0, stream>>>(hv_b, v2t, v2o, M, C, Dv);
    gemm_bf16_nt<8><<<dim3(M / 16, C / 128), 256, 0, stream>>>(hg_b, g2t, gbuf, M, C, Dg);

    // ---- post1: decay / kk-norm / k,v update ----
    post1_kernel<<<M * H, 64, 0, stream>>>(kbuf, vbuf, w2o, a2o, v2o,
                                           v_first, k_k, k_a, a0, v0, w0);

    // ---- WKV7 sequential scan (async double-buffered step vectors) ----
    wkv_kernel<<<Bsz * H, 64, 0, stream>>>(rbuf, w2o, kbuf, vbuf, a2o, v2o, ybuf);

    // ---- post2: GroupNorm + rk residual + gate ----
    post2_kernel<<<M * H, 64, 0, stream>>>(ybuf, rbuf, kbuf, vbuf, gbuf,
                                           r_k, ln_g, ln_b, yg_b);

    // ---- output GEMM (WMMA) ----
    gemm_bf16_nt<8><<<dim3(M / 16, C / 128), 256, 0, stream>>>(yg_b, Wo_b, (float*)d_out, M, C, C);

    // ---- second tuple output: v_first passthrough ----
    (void)hipMemcpyAsync((char*)d_out + MC * sizeof(float), v_first, MC * sizeof(float),
                         hipMemcpyDeviceToDevice, stream);
}
